// CosineContrastiveLoss_61177514164391
// MI455X (gfx1250) — compile-verified, matrix operand order checked by reading, not run on hardware
//
#include <hip/hip_runtime.h>
#include <hip/hip_bf16.h>

// Problem constants (match reference)
#define B_SIZE 4096
#define D_DIM  512
#define BT     128              // block tile (rows == cols)
#define MARGIN_SAME 0.3f
#define MARGIN_DIFF 0.5f
#define POSDIST_MARGIN 0.05f
#define EPS_NORM 1e-8f

#define LDS_STRIDE 40           // 32 K-elements + 8 pad (80B rows: conflict-free frag reads)

typedef __attribute__((ext_vector_type(16))) __bf16 v16bf;
typedef __attribute__((ext_vector_type(8)))  __bf16 v8bf;
typedef __attribute__((ext_vector_type(8)))  float  v8f;
typedef __attribute__((ext_vector_type(4)))  float  v4f;
typedef __attribute__((ext_vector_type(4)))  int    v4i;

// Pointer types for the async global->LDS builtin:
//   param 0: int4 in the global address space, param 1: int4 in LDS.
typedef __attribute__((address_space(1))) v4i* as1_v4i_ptr;
typedef __attribute__((address_space(3))) v4i* as3_v4i_ptr;

// CDNA5 async global->LDS path (ASYNCcnt), with portable fallback.
#if defined(__has_builtin)
#if __has_builtin(__builtin_amdgcn_global_load_async_to_lds_b128)
#define ASYNC_LDS 1
#endif
#endif

#if defined(ASYNC_LDS)
__device__ inline void async_copy16(const __bf16* g, __bf16* l) {
    __builtin_amdgcn_global_load_async_to_lds_b128(
        (as1_v4i_ptr)(void*)g, (as3_v4i_ptr)(void*)l, 0, 0);
}
#endif

// ---------------------------------------------------------------------------
// Zero the scalar output (harness poisons d_out with 0xAA).
// ---------------------------------------------------------------------------
__global__ void init_out_kernel(float* out) {
    if (threadIdx.x == 0 && blockIdx.x == 0) out[0] = 0.0f;
}

// ---------------------------------------------------------------------------
// Pass 1: L2-normalize each 512-float row, store as bf16.
// One wave32 per row; each lane handles 16 contiguous floats.
// ---------------------------------------------------------------------------
__global__ __launch_bounds__(256)
void normalize_rows_kernel(const float* __restrict__ proj,
                           unsigned short* __restrict__ nbf_raw) {
    const int wave = threadIdx.x >> 5;
    const int lane = threadIdx.x & 31;
    const int row  = blockIdx.x * 8 + wave;

    const float* p = proj + (size_t)row * D_DIM + lane * 16;
    v4f a = *(const v4f*)(p);
    v4f b = *(const v4f*)(p + 4);
    v4f c = *(const v4f*)(p + 8);
    v4f d = *(const v4f*)(p + 12);

    float s = 0.0f;
#pragma unroll
    for (int t = 0; t < 4; ++t)
        s += a[t]*a[t] + b[t]*b[t] + c[t]*c[t] + d[t]*d[t];
#pragma unroll
    for (int m = 16; m >= 1; m >>= 1) s += __shfl_xor(s, m, 32);

    const float inv = 1.0f / fmaxf(sqrtf(s), EPS_NORM);

    v8bf o0, o1;
#pragma unroll
    for (int t = 0; t < 4; ++t) {
        o0[t]     = (__bf16)(a[t] * inv);
        o0[t + 4] = (__bf16)(b[t] * inv);
        o1[t]     = (__bf16)(c[t] * inv);
        o1[t + 4] = (__bf16)(d[t] * inv);
    }
    __bf16* q = (__bf16*)nbf_raw + (size_t)row * D_DIM + lane * 16;
    *(v8bf*)(q)     = o0;
    *(v8bf*)(q + 8) = o1;
}

// ---------------------------------------------------------------------------
// Pack per-row key: label in low 16 bits, animacy in high 16 bits.
// ---------------------------------------------------------------------------
__global__ void make_keys_kernel(const int* __restrict__ labels,
                                 const int* __restrict__ class_animacy,
                                 int* __restrict__ keys) {
    const int t = blockIdx.x * blockDim.x + threadIdx.x;
    if (t < B_SIZE) {
        const int l = labels[t];
        keys[t] = (l & 0xffff) | (class_animacy[l] << 16);
    }
}

// ---------------------------------------------------------------------------
// Fragment read from an LDS stripe (row-major, LDS_STRIDE bf16 per row).
// v_wmma 16-bit A/B layout: lanes 0-15 hold row M=lane with K 0..7 and
// 16..23; lanes 16-31 hold the same rows with K 8..15 and 24..31.
// p points at &stripe[localRow * LDS_STRIDE + hi*8].
// ---------------------------------------------------------------------------
__device__ inline v16bf lds_frag(const __bf16* p) {
    v8bf lo = *(const v8bf*)(p);        // K chunk  hi*8 .. +7
    v8bf hh = *(const v8bf*)(p + 16);   // K chunk  16+hi*8 .. +7
    v16bf f;
#pragma unroll
    for (int t = 0; t < 8; ++t) { f[t] = lo[t]; f[t + 8] = hh[t]; }
    return f;
}

// ---------------------------------------------------------------------------
// Pass 2: fused Gram + contrastive loss over upper-triangle 128x128 tiles.
// 8 waves / block. LDS-staged K=32 stripes (A:128x32, B:128x32 bf16), each
// fetched from L2 once per block (async global->LDS when available), then
// re-read by all waves: ~64 FLOP per L2 byte instead of ~16.
// Per wave: 32x64 patch = 2(M) x 4(N) WMMA accumulators.
// ---------------------------------------------------------------------------
__global__ __launch_bounds__(256)
void gram_loss_kernel(const unsigned short* __restrict__ nbf_raw,
                      const int* __restrict__ keys,
                      float* __restrict__ out) {
    const int tileI = blockIdx.y;
    const int tileJ = blockIdx.x;
    if (tileJ < tileI) return;  // uniform per-block: EXEC stays all-ones

    __shared__ __align__(16) __bf16 lA[BT * LDS_STRIDE];
    __shared__ __align__(16) __bf16 lB[BT * LDS_STRIDE];

    const __bf16* nbf = (const __bf16*)nbf_raw;
    const int tid  = threadIdx.x;
    const int wave = tid >> 5;
    const int lane = tid & 31;
    const int l  = lane & 15;
    const int hi = lane >> 4;

    const int rowA0 = tileI * BT;           // global row base of A stripe
    const int colB0 = tileJ * BT;           // global row base of B stripe
    const int wRow  = (wave >> 1) * 32;     // wave's local row offset (0,32,64,96)
    const int wCol  = (wave & 1) * 64;      // wave's local col offset (0,64)

    // Fill slots: 512 16-byte chunks per stripe; each thread owns slots tid, tid+256.
    const int r0 = tid >> 2,  c0 = (tid & 3) * 8;          // slot tid
    const int r1 = (tid + 256) >> 2, c1 = ((tid + 256) & 3) * 8;

    v8f acc[2][4];
#pragma unroll
    for (int wm = 0; wm < 2; ++wm)
#pragma unroll
        for (int wn = 0; wn < 4; ++wn) acc[wm][wn] = (v8f){};

    for (int k0 = 0; k0 < D_DIM; k0 += 32) {
        // ---- stage K-stripes into LDS ----
        const __bf16* gA0 = nbf + (size_t)(rowA0 + r0) * D_DIM + k0 + c0;
        const __bf16* gA1 = nbf + (size_t)(rowA0 + r1) * D_DIM + k0 + c1;
        const __bf16* gB0 = nbf + (size_t)(colB0 + r0) * D_DIM + k0 + c0;
        const __bf16* gB1 = nbf + (size_t)(colB0 + r1) * D_DIM + k0 + c1;
        __bf16* dA0 = &lA[r0 * LDS_STRIDE + c0];
        __bf16* dA1 = &lA[r1 * LDS_STRIDE + c1];
        __bf16* dB0 = &lB[r0 * LDS_STRIDE + c0];
        __bf16* dB1 = &lB[r1 * LDS_STRIDE + c1];
#if defined(ASYNC_LDS)
        async_copy16(gA0, dA0);
        async_copy16(gA1, dA1);
        async_copy16(gB0, dB0);
        async_copy16(gB1, dB1);
#if __has_builtin(__builtin_amdgcn_s_wait_asynccnt)
        __builtin_amdgcn_s_wait_asynccnt(0);
#else
        asm volatile("s_wait_asynccnt 0x0" ::: "memory");
#endif
#else
        *(v8bf*)dA0 = *(const v8bf*)gA0;
        *(v8bf*)dA1 = *(const v8bf*)gA1;
        *(v8bf*)dB0 = *(const v8bf*)gB0;
        *(v8bf*)dB1 = *(const v8bf*)gB1;
#endif
        __syncthreads();

        // ---- compute: 2x4 WMMA accumulators per wave ----
        v16bf a0 = lds_frag(&lA[(wRow      + l) * LDS_STRIDE + hi * 8]);
        v16bf a1 = lds_frag(&lA[(wRow + 16 + l) * LDS_STRIDE + hi * 8]);
#pragma unroll
        for (int wn = 0; wn < 4; ++wn) {
            v16bf b = lds_frag(&lB[(wCol + wn * 16 + l) * LDS_STRIDE + hi * 8]);
            acc[0][wn] = __builtin_amdgcn_wmma_f32_16x16x32_bf16(
                false, a0, false, b, (short)0, acc[0][wn], false, false);
            acc[1][wn] = __builtin_amdgcn_wmma_f32_16x16x32_bf16(
                false, a1, false, b, (short)0, acc[1][wn], false, false);
        }
        __syncthreads();
    }

    // ---- epilogue: C/D layout -> element e of v8f is (M = e + 8*hi, N = l) ----
    float sum = 0.0f;
#pragma unroll
    for (int wm = 0; wm < 2; ++wm) {
        const int iB = rowA0 + wRow + wm * 16 + hi * 8;
        int keyI[8];
#pragma unroll
        for (int e = 0; e < 8; ++e) keyI[e] = keys[iB + e];
#pragma unroll
        for (int wn = 0; wn < 4; ++wn) {
            const int j = colB0 + wCol + wn * 16 + l;
            const int keyJ = keys[j];
            const v8f c = acc[wm][wn];
#pragma unroll
            for (int e = 0; e < 8; ++e) {
                const int i = iB + e;
                if (j > i) {
                    const float d = 1.0f - c[e];
                    const int   x = keyI[e] ^ keyJ;
                    const float margin = ((x >> 16) == 0) ? MARGIN_SAME : MARGIN_DIFF;
                    const float t = ((x & 0xffff) == 0)
                                        ? fmaxf(d - POSDIST_MARGIN, 0.0f)
                                        : fmaxf(margin - d, 0.0f);
                    sum += t * t;
                }
            }
        }
    }

#pragma unroll
    for (int m = 16; m >= 1; m >>= 1) sum += __shfl_xor(sum, m, 32);

    if (lane == 0) {
        const double P = (double)B_SIZE * (double)(B_SIZE - 1) * 0.5;
        atomicAdd(out, sum * (float)(1.0 / P));
    }
}

// ---------------------------------------------------------------------------
// Launcher
// ---------------------------------------------------------------------------
extern "C" void kernel_launch(void* const* d_in, const int* in_sizes, int n_in,
                              void* d_out, int out_size, void* d_ws, size_t ws_size,
                              hipStream_t stream) {
    const float* proj          = (const float*)d_in[0];
    const int*   labels        = (const int*)d_in[1];  // jax default: int64 request -> int32
    const int*   class_animacy = (const int*)d_in[2];
    float*       out           = (float*)d_out;

    // Workspace layout: [bf16 N : 4096*512*2 = 4 MiB][int keys : 16 KiB]
    unsigned short* nbf = (unsigned short*)d_ws;
    int* keys = (int*)((char*)d_ws + (size_t)B_SIZE * D_DIM * sizeof(unsigned short));

    init_out_kernel<<<1, 32, 0, stream>>>(out);
    normalize_rows_kernel<<<B_SIZE / 8, 256, 0, stream>>>(proj, nbf);
    make_keys_kernel<<<B_SIZE / 256, 256, 0, stream>>>(labels, class_animacy, keys);

    dim3 grid(B_SIZE / BT, B_SIZE / BT);  // sub-diagonal tiles exit immediately
    gram_loss_kernel<<<grid, 256, 0, stream>>>(nbf, keys, out);
}
